// ScreenHashLoss_53919019434149
// MI455X (gfx1250) — 8-bit matrix operand order repaired and checked by reading, compiled, not run on hardware
//
#include <hip/hip_runtime.h>

typedef __attribute__((ext_vector_type(8))) int v8i;

#define BDIM 8192
#define LBITS 64

#define ACC_OFF (512 * 1024)

__global__ __launch_bounds__(128) void init_acc_kernel(float* __restrict__ acc) {
    if (threadIdx.x < 67) acc[threadIdx.x] = 0.0f;
}

__global__ __launch_bounds__(256) void prep_kernel(const float* __restrict__ logits,
                                                   const float* __restrict__ bh,
                                                   signed char* __restrict__ bh8,
                                                   float* __restrict__ acc) {
    __shared__ float cs[64];
    __shared__ float qs[256];
    const int tid = threadIdx.x;
    const int i = blockIdx.x * 256 + tid;
    if (tid < 64) cs[tid] = 0.0f;
    __syncthreads();

    const float v = bh[i];
    bh8[i] = (v >= 0.0f) ? (signed char)1 : (signed char)-1;

    const float q = fabsf(fabsf(logits[i]) - 1.0f);
    atomicAdd(&cs[i & 63], v);
    qs[tid] = q;
    __syncthreads();

    for (int s = 128; s > 0; s >>= 1) {
        if (tid < s) qs[tid] += qs[tid + s];
        __syncthreads();
    }
    if (tid == 0) atomicAdd(&acc[1], qs[0]);
    if (tid < 64) atomicAdd(&acc[2 + tid], cs[tid]);
}

__global__ __launch_bounds__(256) void diag_kernel(const float* __restrict__ labels,
                                                   float* __restrict__ acc) {
    __shared__ float ds[256];
    const int tid = threadIdx.x;
    const int i = blockIdx.x * 256 + tid;
    const float lab = labels[(size_t)i * (size_t)(BDIM + 1)];
    float h = fmaxf(fmaf(-lab, 64.0f, 1.0f), 0.0f);
    ds[tid] = h * h;
    __syncthreads();
    for (int s = 128; s > 0; s >>= 1) {
        if (tid < s) ds[tid] += ds[tid + s];
        __syncthreads();
    }
    if (tid == 0) atomicAdd(&acc[66], ds[0]);
}

__global__ __launch_bounds__(256) void sim_kernel(const signed char* __restrict__ bh8,
                                                  const float* __restrict__ labels,
                                                  float* __restrict__ acc) {
    const int lane = threadIdx.x & 31;
    const int wave = threadIdx.x >> 5;
    const int l15  = lane & 15;
    const int hi   = lane >> 4;

    const int baseM  = blockIdx.x * 16;
    const int chunk  = blockIdx.y * 8 + wave;

    v8i a;
    {
        const signed char* arow = bh8 + (baseM + l15) * LBITS + hi * 16;
        const int4 p0 = *(const int4*)(arow);
        const int4 p1 = *(const int4*)(arow + 32);
        a[0] = p0.x; a[1] = p0.y; a[2] = p0.z; a[3] = p0.w;
        a[4] = p1.x; a[5] = p1.y; a[6] = p1.z; a[7] = p1.w;
    }

    float partial = 0.0f;

    for (int t = 0; t < 4; ++t) {
        const int nTile = chunk * 4 + t;
        const int baseN = nTile * 16;

        v8i b;
        {
            const signed char* brow = bh8 + (baseN + l15) * LBITS + hi * 16;
            const int4 q0 = *(const int4*)(brow);
            const int4 q1 = *(const int4*)(brow + 32);
            b[0] = q0.x; b[1] = q0.y; b[2] = q0.z; b[3] = q0.w;
            b[4] = q1.x; b[5] = q1.y; b[6] = q1.z; b[7] = q1.w;
        }

        v8i d = {};
        d = __builtin_amdgcn_wmma_i32_16x16x64_iu8(true, a, true, b, d, false, false);

        const int gcol  = baseN + l15;
        const int rbase = baseM + hi * 8;
#pragma unroll
        for (int r = 0; r < 8; ++r) {
            const int grow = rbase + r;
            const float lab = __builtin_nontemporal_load(
                labels + (size_t)grow * (size_t)BDIM + (size_t)gcol);
            const float g = (float)d[r];
            const float h = fmaxf(fmaf(-lab, g, 1.0f), 0.0f);
            partial = fmaf(h, h, partial);
        }
    }

#pragma unroll
    for (int o = 16; o > 0; o >>= 1)
        partial += __shfl_xor(partial, o, 32);
    if (lane == 0) atomicAdd(&acc[0], partial);
}

__global__ __launch_bounds__(64) void finalize_kernel(const float* __restrict__ acc,
                                                      float* __restrict__ out) {
    __shared__ float bs[64];
    const int tid = threadIdx.x;
    const float m = acc[2 + tid] * (1.0f / (float)BDIM);
    bs[tid] = m * m;
    __syncthreads();
    for (int s = 32; s > 0; s >>= 1) {
        if (tid < s) bs[tid] += bs[tid + s];
        __syncthreads();
    }
    if (tid == 0) {
        const float sim   = (acc[0] - acc[66]) / ((float)BDIM * (float)(BDIM - 1));
        const float quant = acc[1] * (1.0f / ((float)BDIM * (float)LBITS));
        const float bal   = bs[0];
        out[0] = sim + 0.5f * quant + 0.1f * bal;
        out[1] = sim;
        out[2] = quant;
        out[3] = bal;
    }
}

extern "C" void kernel_launch(void* const* d_in, const int* in_sizes, int n_in,
                              void* d_out, int out_size, void* d_ws, size_t ws_size,
                              hipStream_t stream) {
    const float* logits = (const float*)d_in[0];
    const float* bh     = (const float*)d_in[1];
    const float* labels = (const float*)d_in[2];

    signed char* bh8 = (signed char*)d_ws;
    float* acc = (float*)((char*)d_ws + ACC_OFF);
    float* out = (float*)d_out;

    init_acc_kernel<<<1, 128, 0, stream>>>(acc);
    prep_kernel<<<(BDIM * LBITS) / 256, 256, 0, stream>>>(logits, bh, bh8, acc);
    diag_kernel<<<BDIM / 256, 256, 0, stream>>>(labels, acc);
    sim_kernel<<<dim3(BDIM / 16, 16), 256, 0, stream>>>(bh8, labels, acc);
    finalize_kernel<<<1, 64, 0, stream>>>(acc, out);
}
